// LocalSelfAttentionWithGaussianBias_65506841198821
// MI455X (gfx1250) — matrix operand order repaired and checked by reading, not compiled, not run on hardware
//
#include <hip/hip_runtime.h>
#include <hip/hip_bf16.h>
#include <math.h>

// Problem constants (match reference)
#define BB    2
#define LL    4096
#define KK    32
#define DIM   256
#define PE    64
#define CTXD  320           // DIM + PE
#define HEADS 8
#define DH    32
#define INNER 256           // HEADS*DH
#define QT    8             // queries per workgroup
#define ROWS  (QT*KK)       // 256 context rows per workgroup

// LDS strides (elements), chosen so 16B fragment chunks stay 16B-aligned
// (stride*2 bytes multiple of 16) and banks are spread.
#define CTX_S 328
#define W_S   328
#define KV_S  40

typedef __attribute__((ext_vector_type(16))) __bf16 v16bf;
typedef __attribute__((ext_vector_type(4)))  __bf16 v4bf;
typedef __attribute__((ext_vector_type(8)))  float  v8f;

// Native converts: let clang lower fptrunc/fpext to gfx1250 bf16 cvt ops.
__device__ __forceinline__ __bf16 f2bf(float f) { return (__bf16)f; }
__device__ __forceinline__ float  bf2f(__bf16 b) { return (float)b; }

__device__ __forceinline__ void store_bf4(__bf16* p, float a, float b,
                                          float c, float d) {
    v4bf t; t[0] = (__bf16)a; t[1] = (__bf16)b; t[2] = (__bf16)c; t[3] = (__bf16)d;
    *(v4bf*)p = t;           // 8-byte LDS store
}

// A fragment: 16x32 bf16 (MxK). lane: m = lane&15, khalf = lane>>4.
// elems 0..15 = A[m, khalf*16 + 0..15]
__device__ __forceinline__ v16bf load_A(const __bf16* lds, int stride,
                                        int m_off, int k_off, int lane) {
    int m = lane & 15, kh = lane >> 4;
    const __bf16* p0 = lds + (m_off + m) * stride + k_off + kh * 16;
    const __bf16* p1 = p0 + 8;
    v16bf a;
#pragma unroll
    for (int j = 0; j < 8; ++j) a[j] = p0[j];
#pragma unroll
    for (int j = 0; j < 8; ++j) a[8 + j] = p1[j];
    return a;
}

// B fragment: 32x16 bf16 (KxN), source stored TRANSPOSED in LDS as [N][K]
// lane: n = lane&15, kbase = (lane>>4)*16; elems 0..15 = B[kbase+e, n].
__device__ __forceinline__ v16bf load_B(const __bf16* ldsT, int stride,
                                        int n_off, int k_off, int lane) {
    int n = lane & 15, kb = (lane >> 4) * 16;
    const __bf16* p = ldsT + (n_off + n) * stride + k_off + kb;
    v16bf b;
#pragma unroll
    for (int j = 0; j < 16; ++j) b[j] = p[j];
    return b;
}

__device__ __forceinline__ v8f wmma_bf16(v16bf a, v16bf b, v8f c) {
    return __builtin_amdgcn_wmma_f32_16x16x32_bf16(
        false, a, false, b, (short)0, c, false, false);
}

// Dynamic LDS layout: see carve-out below.
#define SMEM_BYTES (((256*CTX_S) + 2*(32*W_S) + 2*(ROWS*KV_S) + (QT*INNER) + (16*CTX_S))*2 \
                    + (256 + 256 + QT*INNER)*4)

__global__ void __launch_bounds__(256, 1)
local_attn_gauss_kernel(const float* __restrict__ x,
                        const int*   __restrict__ topk,
                        const float* __restrict__ rpe,
                        const float* __restrict__ distances,
                        const float* __restrict__ Wq,
                        const float* __restrict__ Wk,
                        const float* __restrict__ Wv,
                        const float* __restrict__ Wout,
                        const float* __restrict__ b_out,
                        const float* __restrict__ log_sigma,
                        float* __restrict__ out) {
    extern __shared__ __align__(16) char smem[];
    __bf16* ctx    = (__bf16*)smem;              // 256 x CTX_S (also WqT/WoutT)
    __bf16* wkT    = ctx    + 256 * CTX_S;       // 32 x W_S
    __bf16* wvT    = wkT    + 32  * W_S;         // 32 x W_S
    __bf16* ktile  = wvT    + 32  * W_S;         // ROWS x KV_S
    __bf16* vtile  = ktile  + ROWS * KV_S;       // ROWS x KV_S
    __bf16* qlds   = vtile  + ROWS * KV_S;       // QT x INNER
    __bf16* xtile  = qlds   + QT * INNER;        // 16 x CTX_S
    float*  scores = (float*)(xtile + 16 * CTX_S);  // 256
    float*  dist2  = scores + 256;                  // 256
    float*  outacc = dist2  + 256;                  // QT*INNER

    const int tid  = threadIdx.x;
    const int lane = tid & 31;
    const int wv_  = tid >> 5;            // wave id, 0..7
    const int b    = blockIdx.x >> 9;     // 1024 blocks = 2 * 512
    const int l0   = (blockIdx.x & 511) * QT;

    // ---------- Stage 1: WqT (into ctx region), x-tile, dist2, zero outacc ----
    // WqT[n][k] = Wq[k][n]; thread t owns row n=t; reads coalesced across t.
#pragma unroll 4
    for (int k4 = 0; k4 < DIM; k4 += 4) {
        store_bf4(ctx + tid * CTX_S + k4,
                  Wq[(size_t)(k4 + 0) * INNER + tid],
                  Wq[(size_t)(k4 + 1) * INNER + tid],
                  Wq[(size_t)(k4 + 2) * INNER + tid],
                  Wq[(size_t)(k4 + 3) * INNER + tid]);
    }
    // x-tile rows 0..7 = queries, rows 8..15 = zero (16x64 float4 units)
#pragma unroll
    for (int j = 0; j < 4; ++j) {
        int u = tid + 256 * j;
        int r = u >> 6, c4 = (u & 63) * 4;
        if (r < QT) {
            float4 v = *(const float4*)(x + ((size_t)b * LL + l0 + r) * DIM + c4);
            store_bf4(xtile + r * CTX_S + c4, v.x, v.y, v.z, v.w);
        } else {
            store_bf4(xtile + r * CTX_S + c4, 0.f, 0.f, 0.f, 0.f);
        }
    }
    for (int i = tid; i < QT * INNER; i += 256) outacc[i] = 0.f;
    {
        int q = tid >> 5, k = tid & 31;
        float d = distances[((size_t)b * LL + l0 + q) * KK + k];
        dist2[tid] = d * d;
    }
    __syncthreads();

    // ---------- Stage 2: Q = x @ Wq  (WMMA, bf16 -> f32) ----------------------
    for (int nt = wv_ * 2; nt < wv_ * 2 + 2; ++nt) {
        v8f acc = {0.f,0.f,0.f,0.f,0.f,0.f,0.f,0.f};
#pragma unroll
        for (int kc = 0; kc < 8; ++kc) {
            v16bf a  = load_A(xtile, CTX_S, 0, kc * 32, lane);
            v16bf bf = load_B(ctx,   CTX_S, nt * 16, kc * 32, lane);
            acc = wmma_bf16(a, bf, acc);
        }
        if (lane < 16) {            // rows 0..7 live in lanes 0..15 (m = r)
            int n = nt * 16 + lane;
#pragma unroll
            for (int r = 0; r < 8; ++r)
                qlds[r * INNER + n] = f2bf(acc[r]);
        }
    }
    __syncthreads();

    // ---------- Stage 3: gather context (overwrites WqT region) ---------------
    // row = q*32+k ; cols 0..255 from x[idx], 256..319 from rpe.
    for (int rr = 0; rr < ROWS / 32; ++rr) {
        int row = rr * 32 + (tid >> 3);
        int sub = tid & 7;
        int q = row >> 5, k = row & 31;
        int idx = topk[((size_t)b * LL + l0 + q) * KK + k];
        const float* xrow = x   + ((size_t)b * LL + idx) * DIM;
        const float* prow = rpe + (((size_t)b * LL + l0 + q) * KK + k) * PE;
#pragma unroll
        for (int cc = 0; cc < 10; ++cc) {
            int c = sub * 4 + cc * 32;
            float4 v = (c < DIM) ? *(const float4*)(xrow + c)
                                 : *(const float4*)(prow + (c - DIM));
            store_bf4(ctx + row * CTX_S + c, v.x, v.y, v.z, v.w);
        }
    }
    __syncthreads();

    // ---------- Stage 4: per-head K/V projection + attention -------------------
    for (int h = 0; h < HEADS; ++h) {
        // stage WkT_h / WvT_h : thread owns contiguous 40-elem k-run of row n
        {
            int n = tid & 31, k0 = (tid >> 5) * 40;
#pragma unroll 2
            for (int i = 0; i < 40; i += 4) {
                int k = k0 + i;
                store_bf4(wkT + n * W_S + k,
                          Wk[(size_t)(k + 0) * INNER + h * DH + n],
                          Wk[(size_t)(k + 1) * INNER + h * DH + n],
                          Wk[(size_t)(k + 2) * INNER + h * DH + n],
                          Wk[(size_t)(k + 3) * INNER + h * DH + n]);
                store_bf4(wvT + n * W_S + k,
                          Wv[(size_t)(k + 0) * INNER + h * DH + n],
                          Wv[(size_t)(k + 1) * INNER + h * DH + n],
                          Wv[(size_t)(k + 2) * INNER + h * DH + n],
                          Wv[(size_t)(k + 3) * INNER + h * DH + n]);
            }
        }
        __syncthreads();

        // K/V projection: each wave does 2 M-tiles of 16 context rows
        for (int mi = 0; mi < 2; ++mi) {
            int mt = wv_ * 2 + mi;
            v8f aK0 = {0.f,0.f,0.f,0.f,0.f,0.f,0.f,0.f};
            v8f aK1 = aK0, aV0 = aK0, aV1 = aK0;
#pragma unroll
            for (int kc = 0; kc < 10; ++kc) {
                v16bf a = load_A(ctx, CTX_S, mt * 16, kc * 32, lane);
                aK0 = wmma_bf16(a, load_B(wkT, W_S,  0, kc * 32, lane), aK0);
                aK1 = wmma_bf16(a, load_B(wkT, W_S, 16, kc * 32, lane), aK1);
                aV0 = wmma_bf16(a, load_B(wvT, W_S,  0, kc * 32, lane), aV0);
                aV1 = wmma_bf16(a, load_B(wvT, W_S, 16, kc * 32, lane), aV1);
            }
            int n0 = lane & 15;
            int mb = mt * 16 + 8 * (lane >> 4);
#pragma unroll
            for (int r = 0; r < 8; ++r) {
                ktile[(mb + r) * KV_S + n0]      = f2bf(aK0[r]);
                ktile[(mb + r) * KV_S + 16 + n0] = f2bf(aK1[r]);
                vtile[(mb + r) * KV_S + n0]      = f2bf(aV0[r]);
                vtile[(mb + r) * KV_S + 16 + n0] = f2bf(aV1[r]);
            }
        }
        __syncthreads();

        // scores[q][k] = scale * <Q_h[q], K_h[q,k]> + gaussian bias
        {
            int q = tid >> 5, k = tid & 31;
            float s = 0.f;
#pragma unroll
            for (int d = 0; d < DH; ++d)
                s += bf2f(qlds[q * INNER + h * DH + d]) *
                     bf2f(ktile[(q * 32 + k) * KV_S + d]);
            float sig = __expf(log_sigma[h]);
            s = s * 0.17677669529663687f - dist2[q * 32 + k] / (2.f * sig * sig);
            scores[q * 32 + k] = s;
        }
        __syncthreads();

        // softmax over k (one thread per query; 32 values each)
        if (tid < QT) {
            float mx = -1e30f;
            for (int k = 0; k < KK; ++k) mx = fmaxf(mx, scores[tid * 32 + k]);
            float sum = 0.f;
            for (int k = 0; k < KK; ++k) {
                float e = __expf(scores[tid * 32 + k] - mx);
                scores[tid * 32 + k] = e; sum += e;
            }
            float inv = 1.f / sum;
            for (int k = 0; k < KK; ++k) scores[tid * 32 + k] *= inv;
        }
        __syncthreads();

        // out_h[q][d] = sum_k P[q,k] * V_h[q,k,d]  (disjoint columns per head)
        {
            int q = tid >> 5, d = tid & 31;
            float o = 0.f;
#pragma unroll
            for (int k = 0; k < KK; ++k)
                o += scores[q * 32 + k] * bf2f(vtile[(q * 32 + k) * KV_S + d]);
            outacc[q * INNER + h * DH + d] = o;
        }
        __syncthreads();
    }

    // ---------- Stage 5: final = outacc @ Wout + b_out  (WMMA) -----------------
    // stage WoutT into ctx region, outacc (bf16) into xtile rows 0..7
#pragma unroll 4
    for (int k4 = 0; k4 < INNER; k4 += 4) {
        store_bf4(ctx + tid * CTX_S + k4,
                  Wout[(size_t)(k4 + 0) * DIM + tid],
                  Wout[(size_t)(k4 + 1) * DIM + tid],
                  Wout[(size_t)(k4 + 2) * DIM + tid],
                  Wout[(size_t)(k4 + 3) * DIM + tid]);
    }
#pragma unroll
    for (int j = 0; j < 2; ++j) {           // 8 rows x 64 float4 units
        int u = tid + 256 * j;
        int r = u >> 6, c4 = (u & 63) * 4;
        float4 v = *(const float4*)(outacc + r * INNER + c4);
        store_bf4(xtile + r * CTX_S + c4, v.x, v.y, v.z, v.w);
    }
    __syncthreads();

    for (int nt = wv_ * 2; nt < wv_ * 2 + 2; ++nt) {
        v8f acc = {0.f,0.f,0.f,0.f,0.f,0.f,0.f,0.f};
#pragma unroll
        for (int kc = 0; kc < 8; ++kc) {
            v16bf a  = load_A(xtile, CTX_S, 0, kc * 32, lane);
            v16bf bf = load_B(ctx,   CTX_S, nt * 16, kc * 32, lane);
            acc = wmma_bf16(a, bf, acc);
        }
        if (lane < 16) {
            int n = nt * 16 + lane;
            float bo = b_out[n];
#pragma unroll
            for (int r = 0; r < 8; ++r)
                out[((size_t)b * LL + l0 + r) * DIM + n] = acc[r] + bo;
        }
    }
}

extern "C" void kernel_launch(void* const* d_in, const int* in_sizes, int n_in,
                              void* d_out, int out_size, void* d_ws, size_t ws_size,
                              hipStream_t stream) {
    (void)in_sizes; (void)n_in; (void)out_size; (void)d_ws; (void)ws_size;
    const float* x         = (const float*)d_in[0];
    const int*   topk      = (const int*)  d_in[1];
    const float* rpe       = (const float*)d_in[2];
    const float* distances = (const float*)d_in[3];
    const float* Wq        = (const float*)d_in[4];
    const float* Wk        = (const float*)d_in[5];
    const float* Wv        = (const float*)d_in[6];
    const float* Wout      = (const float*)d_in[7];
    const float* b_out     = (const float*)d_in[8];
    const float* log_sigma = (const float*)d_in[9];
    float* out = (float*)d_out;

    dim3 grid(BB * (LL / QT));   // 1024 workgroups
    dim3 block(256);             // 8 wave32 waves
    local_attn_gauss_kernel<<<grid, block, SMEM_BYTES, stream>>>(
        x, topk, rpe, distances, Wq, Wk, Wv, Wout, b_out, log_sigma, out);
}
